// MambaBlock_39238821216757
// MI455X (gfx1250) — matrix operand order repaired and checked by reading, not compiled, not run on hardware
//
#include <hip/hip_runtime.h>
#include <hip/hip_bf16.h>
#include <math.h>

// ---------------------------------------------------------------------------
// Mamba block for MI455X (gfx1250): bf16 WMMA GEMMs (async-LDS double-buffered)
// + fp32 scan/glue.
// ---------------------------------------------------------------------------

typedef __attribute__((ext_vector_type(16))) __bf16 bf16x16;
typedef __attribute__((ext_vector_type(8)))  __bf16 bf16x8;
typedef __attribute__((ext_vector_type(4)))  __bf16 bf16x4;
typedef __attribute__((ext_vector_type(8)))  float  floatx8;
typedef __attribute__((__vector_size__(16))) int    v4i;   // builtin's pointee type

#define D_MODEL 1024
#define D_STATE 16
#define D_CONV  4
#define D_INNER 2048
#define BATCH   2
#define LENGTH  2048
#define MROWS   (BATCH * LENGTH)          // 4096
#define NPROJ   (D_INNER + 2 * D_STATE)   // 2080

#if defined(__AMDGCN__)
#define HAVE_ASYNC 1
#else
#define HAVE_ASYNC 0
#endif

// Async global -> LDS 16-byte copy (gfx1250 GLOBAL_LOAD_ASYNC_TO_LDS_B128,
// tracked with ASYNCcnt). Signature (probe-confirmed via clang diagnostic):
//   (v4i addrspace(1)* gsrc, v4i addrspace(3)* lds, imm int offset, imm int cpol)
__device__ __forceinline__ void async_load16(const __bf16* g, __bf16* l) {
#if defined(__AMDGCN__)
#if __has_builtin(__builtin_amdgcn_global_load_async_to_lds_b128)
  __builtin_amdgcn_global_load_async_to_lds_b128(
      (__attribute__((address_space(1))) v4i*)g,
      (__attribute__((address_space(3))) v4i*)l, 0, 0);
#else
  asm volatile("global_load_async_to_lds_b128 %0, %1, off"
               :: "v"((unsigned)(uintptr_t)(__attribute__((address_space(3))) void*)l),
                  "v"(g)
               : "memory");
#endif
#else
  (void)g; (void)l;
#endif
}

__device__ __forceinline__ void wait_async0() {
#if defined(__AMDGCN__)
#if __has_builtin(__builtin_amdgcn_s_wait_asynccnt)
  __builtin_amdgcn_s_wait_asynccnt(0);
#else
  asm volatile("s_wait_asynccnt 0x0" ::: "memory");
#endif
#endif
}

// ---------------------------------------------------------------------------
// fp32 -> bf16 cast (strided source, compact dest), 4 elements / thread
// ---------------------------------------------------------------------------
__global__ __launch_bounds__(256) void cast_f32_bf16(
    const float* __restrict__ src, __bf16* __restrict__ dst,
    int rows, int cols, int src_ld)
{
  int idx = blockIdx.x * 256 + threadIdx.x;
  int cpr = cols >> 2;
  int total4 = rows * cpr;
  if (idx >= total4) return;
  int r  = idx / cpr;
  int c4 = (idx - r * cpr) << 2;
  const float4 v = *(const float4*)(src + (size_t)r * src_ld + c4);
  bf16x4 o;
  o.x = (__bf16)v.x; o.y = (__bf16)v.y; o.z = (__bf16)v.z; o.w = (__bf16)v.w;
  *(bf16x4*)(dst + (size_t)r * cols + c4) = o;
}

__device__ __forceinline__ float softplusf(float x) {
  return (x > 20.0f) ? x : log1pf(__expf(x));
}
__device__ __forceinline__ float siluf(float x) {
  return x * (1.0f / (1.0f + __expf(-x)));
}

// ---------------------------------------------------------------------------
// bf16 WMMA GEMM: C[M,N] = A[M,K] * B[N,K]^T  (both row-major, K contiguous)
// Block tile 128x128, BK=32, 8 waves (4m x 2n), wave tile 32x64 (2x4 WMMA).
// Double-buffered LDS; tiles fetched with GLOBAL_LOAD_ASYNC_TO_LDS_B128 and
// consumed next iteration (s_wait_asynccnt + barrier). GUARD=true adds the
// ragged-N bounds checks (only needed for the N=2080 GEMM).
// Epilogue: optional bias[n], softplus, residual add.
// LDS rows padded to 40 bf16 (80B) -> conflict-free 16-lane b128 reads.
// ---------------------------------------------------------------------------
template<bool GUARD>
__global__ __launch_bounds__(256) void gemm_bf16_wmma(
    const __bf16* __restrict__ A, int lda,
    const __bf16* __restrict__ Bw, int ldb,
    float* __restrict__ C, int ldc,
    int M, int N, int K,
    const float* __restrict__ bias,
    const float* __restrict__ resid, int ldr,
    int act)
{
  constexpr int LDST = 40;
  constexpr int TSZ  = 128 * LDST;
  __shared__ __bf16 As[2 * TSZ];
  __shared__ __bf16 Bs[2 * TSZ];

  const int tid  = threadIdx.x;
  const int m0   = blockIdx.y * 128;
  const int n0   = blockIdx.x * 128;
  const int wave = tid >> 5;
  const int lane = tid & 31;
  const int wm   = (wave >> 1) * 32;   // 0,32,64,96
  const int wn   = (wave & 1) * 64;    // 0,64
  const int row  = lane & 15;
  const int half = lane >> 4;
  // tile-load coords: thread covers rows r0 and r0+64 at 8-elem column k8
  const int r0 = tid >> 2;
  const int k8 = (tid & 3) << 3;

  floatx8 acc[2][4];
  const floatx8 zero = {};
#pragma unroll
  for (int i = 0; i < 2; ++i)
#pragma unroll
    for (int j = 0; j < 4; ++j) acc[i][j] = zero;

  auto issue_tile = [&](int buf, int k0) {
#pragma unroll
    for (int h = 0; h < 2; ++h) {
      const int r = r0 + h * 64;
      const int rg = m0 + r;
      if (!GUARD || rg < M) {
        async_load16(A + (size_t)rg * lda + k0 + k8, &As[buf * TSZ + r * LDST + k8]);
      } else {
        const bf16x8 z = {};
        *(bf16x8*)&As[buf * TSZ + r * LDST + k8] = z;
      }
      const int ng = n0 + r;
      if (!GUARD || ng < N) {
        async_load16(Bw + (size_t)ng * ldb + k0 + k8, &Bs[buf * TSZ + r * LDST + k8]);
      } else {
        const bf16x8 z = {};
        *(bf16x8*)&Bs[buf * TSZ + r * LDST + k8] = z;
      }
    }
  };

  auto compute_tile = [&](int buf) {
    bf16x16 af[2], bfr[4];
#pragma unroll
    for (int i = 0; i < 2; ++i) {
      // A (ISA 7.12.2): lane holds row (lane&15); K {0..7,16..23}/{8..15,24..31}
      const __bf16* p = &As[buf * TSZ + (wm + i * 16 + row) * LDST + half * 8];
      bf16x8 lo = *(const bf16x8*)p;
      bf16x8 hi = *(const bf16x8*)(p + 16);
      af[i] = __builtin_shufflevector(lo, hi, 0,1,2,3,4,5,6,7,8,9,10,11,12,13,14,15);
    }
#pragma unroll
    for (int j = 0; j < 4; ++j) {
      // B: lane holds column n=(lane&15) = W row; K {0..7,16..23}/{8..15,24..31}
      const __bf16* q = &Bs[buf * TSZ + (wn + j * 16 + row) * LDST + half * 8];
      bf16x8 lo = *(const bf16x8*)q;
      bf16x8 hi = *(const bf16x8*)(q + 16);
      bfr[j] = __builtin_shufflevector(lo, hi, 0,1,2,3,4,5,6,7,8,9,10,11,12,13,14,15);
    }
#pragma unroll
    for (int i = 0; i < 2; ++i)
#pragma unroll
      for (int j = 0; j < 4; ++j)
        acc[i][j] = __builtin_amdgcn_wmma_f32_16x16x32_bf16(
            false, af[i], false, bfr[j], (short)0, acc[i][j], false, false);
  };

#if HAVE_ASYNC
  // -------- async double-buffered pipeline --------
  const int niter = K / 32;
  issue_tile(0, 0);
  for (int it = 0; it < niter; ++it) {
    const int cur = it & 1;
    wait_async0();       // this wave's async tile (buf cur) landed in LDS
    __syncthreads();     // all waves' tiles landed; prior reads of cur^1 done
    if (it + 1 < niter) issue_tile(cur ^ 1, (it + 1) * 32);
    compute_tile(cur);
  }
#else
  // -------- fallback: synchronous single-buffer tiles (host pass / no async)
  for (int k0 = 0; k0 < K; k0 += 32) {
#pragma unroll
    for (int h = 0; h < 2; ++h) {
      const int r = r0 + h * 64;
      bf16x8 va = {};
      const int rg = m0 + r;
      if (!GUARD || rg < M) va = *(const bf16x8*)(A + (size_t)rg * lda + k0 + k8);
      *(bf16x8*)&As[r * LDST + k8] = va;
      bf16x8 vb = {};
      const int ng = n0 + r;
      if (!GUARD || ng < N) vb = *(const bf16x8*)(Bw + (size_t)ng * ldb + k0 + k8);
      *(bf16x8*)&Bs[r * LDST + k8] = vb;
    }
    __syncthreads();
    compute_tile(0);
    __syncthreads();
  }
#endif

  // ---- epilogue: elem e -> (m = e + half*8, n = lane&15) within 16x16 ----
#pragma unroll
  for (int i = 0; i < 2; ++i) {
#pragma unroll
    for (int j = 0; j < 4; ++j) {
      int n = n0 + wn + j * 16 + row;
      if (!GUARD || n < N) {
        float bv = bias ? bias[n] : 0.0f;
#pragma unroll
        for (int e = 0; e < 8; ++e) {
          int m = m0 + wm + i * 16 + half * 8 + e;
          float v = acc[i][j][e] + bv;
          if (act == 1) v = softplusf(v);
          if (resid)    v += resid[(size_t)m * ldr + n];
          C[(size_t)m * ldc + n] = v;
        }
      }
    }
  }
}

// ---------------------------------------------------------------------------
// Depthwise causal conv1d (4 taps) + SiLU, 4 channels / thread.
// xz is (MROWS, 2*D_INNER); xs = first D_INNER columns.
// ---------------------------------------------------------------------------
__global__ __launch_bounds__(256) void conv_silu_kernel(
    const float* __restrict__ xz,
    const float* __restrict__ conv_w,   // (D_INNER, 1, 4)
    const float* __restrict__ conv_b,
    float* __restrict__ u, __bf16* __restrict__ ub)
{
  int id = blockIdx.x * 256 + threadIdx.x;          // over MROWS * (D_INNER/4)
  if (id >= MROWS * (D_INNER / 4)) return;
  int d4 = id & (D_INNER / 4 - 1);
  int ml = id >> 9;                                 // D_INNER/4 == 512
  int l  = ml & (LENGTH - 1);
  int d  = d4 << 2;
  const float* wr = conv_w + d * 4;
  float a0 = conv_b[d],     a1 = conv_b[d + 1];
  float a2 = conv_b[d + 2], a3 = conv_b[d + 3];
#pragma unroll
  for (int j = 0; j < D_CONV; ++j) {
    int t = l - (D_CONV - 1) + j;
    if (t >= 0) {
      const float4 xv = *(const float4*)(xz + (size_t)(ml - (D_CONV - 1) + j) * (2 * D_INNER) + d);
      a0 += wr[0 * 4 + j] * xv.x;
      a1 += wr[1 * 4 + j] * xv.y;
      a2 += wr[2 * 4 + j] * xv.z;
      a3 += wr[3 * 4 + j] * xv.w;
    }
  }
  a0 = siluf(a0); a1 = siluf(a1); a2 = siluf(a2); a3 = siluf(a3);
  float4 o = {a0, a1, a2, a3};
  *(float4*)(u + (size_t)ml * D_INNER + d) = o;
  bf16x4 ob;
  ob.x = (__bf16)a0; ob.y = (__bf16)a1; ob.z = (__bf16)a2; ob.w = (__bf16)a3;
  *(bf16x4*)(ub + (size_t)ml * D_INNER + d) = ob;
}

// ---------------------------------------------------------------------------
// Selective scan: lane = (b, d, n); 16-lane shuffle-tree reduce over n.
// Fuses +u*D, SiLU(z) gate, bf16 cast of y for the output GEMM.
// 65536 lanes = 2048 wave32s; prefetch 16 steps ahead to hide strided loads.
// ---------------------------------------------------------------------------
__global__ __launch_bounds__(256) void scan_kernel(
    const float* __restrict__ delta,   // (MROWS, D_INNER)
    const float* __restrict__ A_log,   // (D_INNER, D_STATE)
    const float* __restrict__ proj,    // (MROWS, NPROJ): B @ 2048, C @ 2064
    const float* __restrict__ u,       // (MROWS, D_INNER)
    const float* __restrict__ Dp,      // (D_INNER)
    const float* __restrict__ xz,      // (MROWS, 2*D_INNER): z @ D_INNER
    __bf16* __restrict__ yb)           // (MROWS, D_INNER)
{
  int gid = blockIdx.x * 256 + threadIdx.x;   // BATCH*D_INNER*D_STATE = 65536
  int n  = gid & 15;
  int ch = gid >> 4;
  int d  = ch & (D_INNER - 1);
  int b  = ch >> 11;

  const float Aval = -__expf(A_log[d * D_STATE + n]);
  const float Dd   = Dp[d];
  const size_t base = (size_t)b * LENGTH;
  const float* dptr = delta + base * D_INNER + d;
  const float* uptr = u     + base * D_INNER + d;
  const float* Bptr = proj  + base * NPROJ + D_INNER + n;
  const float* Cptr = proj  + base * NPROJ + D_INNER + D_STATE + n;
  const float* zptr = xz    + base * (2 * D_INNER) + D_INNER + d;
  __bf16*      yptr = yb    + base * D_INNER + d;

  float state = 0.0f;
  for (int t = 0; t < LENGTH; ++t) {
    if (t + 16 < LENGTH) {   // hide strided-load latency
      __builtin_prefetch(dptr + (size_t)(t + 16) * D_INNER, 0, 0);
      __builtin_prefetch(uptr + (size_t)(t + 16) * D_INNER, 0, 0);
      __builtin_prefetch(Bptr + (size_t)(t + 16) * NPROJ,  0, 0);
    }
    float dt = dptr[(size_t)t * D_INNER];
    float uu = uptr[(size_t)t * D_INNER];
    float Bt = Bptr[(size_t)t * NPROJ];
    float Ct = Cptr[(size_t)t * NPROJ];
    state = state * __expf(dt * Aval) + (dt * Bt) * uu;
    float y = state * Ct;
    y += __shfl_xor(y, 1, 32);
    y += __shfl_xor(y, 2, 32);
    y += __shfl_xor(y, 4, 32);
    y += __shfl_xor(y, 8, 32);
    if (n == 0) {
      float z   = zptr[(size_t)t * (2 * D_INNER)];
      float out = (y + uu * Dd) * siluf(z);
      yptr[(size_t)t * D_INNER] = (__bf16)out;
    }
  }
}

// ---------------------------------------------------------------------------
// LayerNorm over D_MODEL=1024: one 256-thread block per row, float4 lanes.
// ---------------------------------------------------------------------------
__global__ __launch_bounds__(256) void layernorm_kernel(
    const float* __restrict__ in, const float* __restrict__ g,
    const float* __restrict__ bb, float* __restrict__ out)
{
  __shared__ float red[16];
  int rowi = blockIdx.x;
  int tid  = threadIdx.x;
  const float4 v = *(const float4*)(in + (size_t)rowi * D_MODEL + tid * 4);
  float s  = v.x + v.y + v.z + v.w;
  float s2 = v.x * v.x + v.y * v.y + v.z * v.z + v.w * v.w;
#pragma unroll
  for (int off = 16; off; off >>= 1) {
    s  += __shfl_xor(s,  off, 32);
    s2 += __shfl_xor(s2, off, 32);
  }
  int wave = tid >> 5, lane = tid & 31;
  if (lane == 0) { red[wave] = s; red[8 + wave] = s2; }
  __syncthreads();
  if (tid == 0) {
    float ts = 0.f, t2 = 0.f;
    for (int w = 0; w < 8; ++w) { ts += red[w]; t2 += red[8 + w]; }
    red[0] = ts; red[8] = t2;
  }
  __syncthreads();
  float mu  = red[0] * (1.0f / D_MODEL);
  float var = red[8] * (1.0f / D_MODEL) - mu * mu;
  float inv = rsqrtf(var + 1e-5f);
  const float4 gv = *(const float4*)(g  + tid * 4);
  const float4 bv = *(const float4*)(bb + tid * 4);
  float4 o;
  o.x = (v.x - mu) * inv * gv.x + bv.x;
  o.y = (v.y - mu) * inv * gv.y + bv.y;
  o.z = (v.z - mu) * inv * gv.z + bv.z;
  o.w = (v.w - mu) * inv * gv.w + bv.w;
  *(float4*)(out + (size_t)rowi * D_MODEL + tid * 4) = o;
}

// ---------------------------------------------------------------------------
extern "C" void kernel_launch(void* const* d_in, const int* in_sizes, int n_in,
                              void* d_out, int out_size, void* d_ws, size_t ws_size,
                              hipStream_t stream) {
  (void)in_sizes; (void)n_in; (void)out_size; (void)ws_size;
  const float* x     = (const float*)d_in[0];
  const float* W_in  = (const float*)d_in[1];
  const float* convw = (const float*)d_in[2];
  const float* convb = (const float*)d_in[3];
  const float* W_x   = (const float*)d_in[4];
  const float* W_dt  = (const float*)d_in[5];
  const float* b_dt  = (const float*)d_in[6];
  const float* A_log = (const float*)d_in[7];
  const float* Dp    = (const float*)d_in[8];
  const float* W_out = (const float*)d_in[9];
  const float* ln_g  = (const float*)d_in[10];
  const float* ln_b  = (const float*)d_in[11];
  float* out = (float*)d_out;

  char* w = (char*)d_ws;
  auto alloc = [&](size_t bytes) -> char* {
    char* p = w;
    w += (bytes + 255) & ~(size_t)255;
    return p;
  };
  __bf16* Wb_in  = (__bf16*)alloc((size_t)(2 * D_INNER) * D_MODEL * 2);
  __bf16* Wb_x   = (__bf16*)alloc((size_t)NPROJ * D_INNER * 2);
  __bf16* Wb_dt  = (__bf16*)alloc((size_t)D_INNER * D_INNER * 2);
  __bf16* Wb_out = (__bf16*)alloc((size_t)D_MODEL * D_INNER * 2);
  __bf16* xb     = (__bf16*)alloc((size_t)MROWS * D_MODEL * 2);
  float*  xz     = (float*) alloc((size_t)MROWS * 2 * D_INNER * 4);
  float*  u      = (float*) alloc((size_t)MROWS * D_INNER * 4);
  __bf16* ub     = (__bf16*)alloc((size_t)MROWS * D_INNER * 2);
  float*  proj   = (float*) alloc((size_t)MROWS * NPROJ * 4);
  __bf16* dinb   = (__bf16*)alloc((size_t)MROWS * D_INNER * 2);
  float*  delta  = (float*) alloc((size_t)MROWS * D_INNER * 4);
  __bf16* yb     = (__bf16*)alloc((size_t)MROWS * D_INNER * 2);
  float*  outp   = (float*) alloc((size_t)MROWS * D_MODEL * 4);

  dim3 blk(256);
  auto cgrid = [](int rows, int cols) { return dim3((rows * (cols / 4) + 255) / 256); };

  // bf16 casts of weights + input
  cast_f32_bf16<<<cgrid(2 * D_INNER, D_MODEL), blk, 0, stream>>>(W_in,  Wb_in,  2 * D_INNER, D_MODEL, D_MODEL);
  cast_f32_bf16<<<cgrid(NPROJ, D_INNER),       blk, 0, stream>>>(W_x,   Wb_x,   NPROJ, D_INNER, D_INNER);
  cast_f32_bf16<<<cgrid(D_INNER, D_INNER),     blk, 0, stream>>>(W_dt,  Wb_dt,  D_INNER, D_INNER, D_INNER);
  cast_f32_bf16<<<cgrid(D_MODEL, D_INNER),     blk, 0, stream>>>(W_out, Wb_out, D_MODEL, D_INNER, D_INNER);
  cast_f32_bf16<<<cgrid(MROWS, D_MODEL),       blk, 0, stream>>>(x,     xb,     MROWS, D_MODEL, D_MODEL);

  // GEMM1: xz = xb (4096x1024) @ W_in^T (4096x1024)  -> (4096, 4096)
  gemm_bf16_wmma<false><<<dim3(32, 32), blk, 0, stream>>>(
      xb, D_MODEL, Wb_in, D_MODEL, xz, 2 * D_INNER,
      MROWS, 2 * D_INNER, D_MODEL, nullptr, nullptr, 0, 0);

  // depthwise causal conv + SiLU -> u (f32) + ub (bf16)
  conv_silu_kernel<<<dim3((MROWS * (D_INNER / 4)) / 256), blk, 0, stream>>>(
      xz, convw, convb, u, ub);

  // GEMM2: proj = ub (4096x2048) @ W_x^T (2080x2048) -> (4096, 2080)
  gemm_bf16_wmma<true><<<dim3((NPROJ + 127) / 128, 32), blk, 0, stream>>>(
      ub, D_INNER, Wb_x, D_INNER, proj, NPROJ,
      MROWS, NPROJ, D_INNER, nullptr, nullptr, 0, 0);

  // cast d_in slice (stride NPROJ) to compact bf16
  cast_f32_bf16<<<cgrid(MROWS, D_INNER), blk, 0, stream>>>(
      proj, dinb, MROWS, D_INNER, NPROJ);

  // GEMM3: delta = softplus(dinb @ W_dt^T + b_dt) -> (4096, 2048)
  gemm_bf16_wmma<false><<<dim3(16, 32), blk, 0, stream>>>(
      dinb, D_INNER, Wb_dt, D_INNER, delta, D_INNER,
      MROWS, D_INNER, D_INNER, b_dt, nullptr, 0, 1);

  // selective scan + gate -> yb (bf16)
  scan_kernel<<<dim3((BATCH * D_INNER * D_STATE) / 256), blk, 0, stream>>>(
      delta, A_log, proj, u, Dp, xz, yb);

  // GEMM4: outp = yb @ W_out^T + x -> (4096, 1024)
  gemm_bf16_wmma<false><<<dim3(8, 32), blk, 0, stream>>>(
      yb, D_INNER, Wb_out, D_INNER, outp, D_MODEL,
      MROWS, D_MODEL, D_INNER, nullptr, x, D_MODEL, 0);

  // layernorm -> d_out
  layernorm_kernel<<<dim3(MROWS), blk, 0, stream>>>(outp, ln_g, ln_b, out);
}
